// CoMPT_7516192768954
// MI455X (gfx1250) — matrix operand order repaired and checked by reading, not compiled, not run on hardware
//
#include <hip/hip_runtime.h>
#include <hip/hip_bf16.h>

// CoMPT graph transformer, B=8,N=128,D=512,H=16,DK=32.
// Big GEMMs (k = key_edge@Wk, e = edge_hidden@We, 68.7 GFLOP each) run on the
// WMMA pipe in f16 with f32 accumulation (v_wmma_f32_16x16x32_f16), with
// double-buffered LDS and register-staged global prefetch.
// ws layout (bytes), total ~167.3 MB:
//   WtK f16 512K | WtE f16 512K | q f32 2M | v f32 2M | adjs 512K | nhx 2M
//   outS 8M | inS 8M | message 8M | kf16 134M

typedef _Float16 v16h __attribute__((ext_vector_type(16)));
typedef _Float16 h8   __attribute__((ext_vector_type(8)));
typedef _Float16 h4   __attribute__((ext_vector_type(4)));
typedef float    v8f  __attribute__((ext_vector_type(8)));

#define Bn 8
#define Nn 128
#define Dn 512
#define Hn 16
#define DKn 32

// ---------------- small helpers ----------------
__device__ __forceinline__ float blockMax128(float v, float* red, int tid) {
    red[tid] = v; __syncthreads();
    for (int s = 64; s > 0; s >>= 1) {
        if (tid < s) red[tid] = fmaxf(red[tid], red[tid + s]);
        __syncthreads();
    }
    float r = red[0]; __syncthreads();
    return r;
}
__device__ __forceinline__ float blockSum128(float v, float* red, int tid) {
    red[tid] = v; __syncthreads();
    for (int s = 64; s > 0; s >>= 1) {
        if (tid < s) red[tid] = red[tid] + red[tid + s];
        __syncthreads();
    }
    float r = red[0]; __syncthreads();
    return r;
}

// ---------------- prep: W transpose + f16 convert ----------------
// Wt[n][k] = (f16) W[k][n]  so that B-fragment K runs are contiguous.
__global__ void wconv_kernel(const float* __restrict__ W, _Float16* __restrict__ Wt) {
    int k = blockIdx.x * 16 + threadIdx.x;
    int n = blockIdx.y * 16 + threadIdx.y;
    Wt[(size_t)n * Dn + k] = (_Float16)W[(size_t)k * Dn + n];
}

// ---------------- prep: q/v projection, head-major output ----------------
__global__ void qv_proj_kernel(const float* __restrict__ X, const float* __restrict__ W,
                               const float* __restrict__ bias, float* __restrict__ out) {
    __shared__ float row[Dn];
    int rid = blockIdx.x;           // b*N + m
    int tid = threadIdx.x;          // 256
    row[tid]       = X[(size_t)rid * Dn + tid];
    row[tid + 256] = X[(size_t)rid * Dn + tid + 256];
    __syncthreads();
    int b = rid >> 7, m = rid & 127;
    for (int cc = 0; cc < 2; ++cc) {
        int c = tid + cc * 256;
        float acc = bias[c];
        for (int k = 0; k < Dn; ++k) acc += row[k] * W[(size_t)k * Dn + c];
        int h = c >> 5, dk = c & 31;
        out[(((size_t)b * Hn + h) * Nn + m) * DKn + dk] = acc;
    }
}

// ---------------- prep: adj softmax ----------------
__global__ void adj_softmax_kernel(const float* __restrict__ adj, const int* __restrict__ mask,
                                   const float* __restrict__ lam, float* __restrict__ adjs) {
    __shared__ float red[Nn];
    int i = blockIdx.x, b = blockIdx.y, j = threadIdx.x;
    float l = lam[0];
    float v = (mask[b * Nn + j] == 0) ? -__builtin_inff()
                                      : (-l * adj[((size_t)b * Nn + i) * Nn + j]);
    float mx = blockMax128(v, red, j);
    float e  = __expf(v - mx);
    float s  = blockSum128(e, red, j);
    adjs[((size_t)b * Nn + i) * Nn + j] = e / s;
}

// ---------------- WMMA fragment loaders ----------------
// A 16x32 f16 (ISA table): lane r=lane&15 is row M; half=lane>>4.
// elems 0..7  = K[kk + 8*half + 0..7], elems 8..15 = K[kk + 8*half + 16..23].
__device__ __forceinline__ v16h load_afrag(const _Float16* p) {
    h8 lo = *(const h8*)p;
    h8 hi = *(const h8*)(p + 16);
    return __builtin_shufflevector(lo, hi, 0,1,2,3,4,5,6,7,8,9,10,11,12,13,14,15);
}
// B 32x16 f16: lane r is col N; half=lane>>4.
// elems 0..7  = K[kk + 8*half + 0..7], elems 8..15 = K[kk + 8*half + 16..23].
__device__ __forceinline__ v16h load_bfrag(const _Float16* p) {
    h8 lo = *(const h8*)p;
    h8 hi = *(const h8*)(p + 16);
    return __builtin_shufflevector(lo, hi, 0,1,2,3,4,5,6,7,8,9,10,11,12,13,14,15);
}

// ---------------- big GEMM 1: k = key_edge @ Wk + bk -> kf16 ----------------
// grid (4, N, B). 8 waves; wave w owns rows 16w..16w+15, 8 N-tiles.
// Double-buffered LDS K-chunks of 64; next chunk prefetched to regs pre-compute.
__global__ __launch_bounds__(256) void k_gemm_kernel(const float* __restrict__ key_edge,
                                                     const _Float16* __restrict__ WtK,
                                                     const float* __restrict__ bk,
                                                     _Float16* __restrict__ kf16) {
    __shared__ __align__(16) _Float16 sA[2][Nn * 64];
    __shared__ __align__(16) _Float16 sB[2][Nn * 64];
    int nb = blockIdx.x, m = blockIdx.y, b = blockIdx.z;
    int tid = threadIdx.x;
    int w = tid >> 5, lane = tid & 31, r = lane & 15, half = lane >> 4;

    v8f acc[8] = {};
    const float*    Abase = key_edge + ((size_t)(b * Nn + m)) * Nn * Dn;
    const _Float16* Bbase = WtK + (size_t)nb * 128 * Dn;

    float4 aReg[8];
    h8     bReg[4];

    // prologue: load + stage chunk 0
#pragma unroll
    for (int i = 0; i < 8; ++i) {
        int idx = tid + i * 256, n = idx >> 4, k4 = (idx & 15) << 2;
        aReg[i] = *(const float4*)(Abase + (size_t)n * Dn + k4);
    }
#pragma unroll
    for (int i = 0; i < 4; ++i) {
        int idx = tid + i * 256, n = idx >> 3, k8 = (idx & 7) << 3;
        bReg[i] = *(const h8*)(Bbase + (size_t)n * Dn + k8);
    }
#pragma unroll
    for (int i = 0; i < 8; ++i) {
        int idx = tid + i * 256, n = idx >> 4, k4 = (idx & 15) << 2;
        h4 hv = { (_Float16)aReg[i].x, (_Float16)aReg[i].y,
                  (_Float16)aReg[i].z, (_Float16)aReg[i].w };
        *(h4*)&sA[0][n * 64 + k4] = hv;
    }
#pragma unroll
    for (int i = 0; i < 4; ++i) {
        int idx = tid + i * 256, n = idx >> 3, k8 = (idx & 7) << 3;
        *(h8*)&sB[0][n * 64 + k8] = bReg[i];
    }
    __syncthreads();

    int buf = 0;
    for (int c = 0; c < 8; ++c) {
        int kn = (c + 1) * 64;
        if (c < 7) {  // issue next-chunk global loads before compute
#pragma unroll
            for (int i = 0; i < 8; ++i) {
                int idx = tid + i * 256, n = idx >> 4, k4 = (idx & 15) << 2;
                aReg[i] = *(const float4*)(Abase + (size_t)n * Dn + kn + k4);
            }
#pragma unroll
            for (int i = 0; i < 4; ++i) {
                int idx = tid + i * 256, n = idx >> 3, k8 = (idx & 7) << 3;
                bReg[i] = *(const h8*)(Bbase + (size_t)n * Dn + kn + k8);
            }
        }
        // compute on current buffer; batch B-fragment loads in groups of 4
#pragma unroll
        for (int kk = 0; kk < 64; kk += 32) {
            v16h af = load_afrag(&sA[buf][(16 * w + r) * 64 + kk + half * 8]);
#pragma unroll
            for (int g = 0; g < 2; ++g) {
                v16h bf[4];
#pragma unroll
                for (int t = 0; t < 4; ++t)
                    bf[t] = load_bfrag(&sB[buf][(16 * (g * 4 + t) + r) * 64 + kk + half * 8]);
#pragma unroll
                for (int t = 0; t < 4; ++t)
                    acc[g * 4 + t] = __builtin_amdgcn_wmma_f32_16x16x32_f16(
                        false, af, false, bf[t], (short)0, acc[g * 4 + t], false, false);
            }
        }
        if (c < 7) {  // stage next chunk into alternate buffer
#pragma unroll
            for (int i = 0; i < 8; ++i) {
                int idx = tid + i * 256, n = idx >> 4, k4 = (idx & 15) << 2;
                h4 hv = { (_Float16)aReg[i].x, (_Float16)aReg[i].y,
                          (_Float16)aReg[i].z, (_Float16)aReg[i].w };
                *(h4*)&sA[buf ^ 1][n * 64 + k4] = hv;
            }
#pragma unroll
            for (int i = 0; i < 4; ++i) {
                int idx = tid + i * 256, n = idx >> 3, k8 = (idx & 7) << 3;
                *(h8*)&sB[buf ^ 1][n * 64 + k8] = bReg[i];
            }
        }
        __syncthreads();
        buf ^= 1;
    }
    // epilogue: + bk, store f16 k tile. C layout: VGPR vr -> row 16w+vr+8*half, col lane&15.
#pragma unroll
    for (int nt = 0; nt < 8; ++nt) {
        int col = nb * 128 + nt * 16 + r;
        float bkc = bk[col];
#pragma unroll
        for (int vr = 0; vr < 8; ++vr) {
            int n = 16 * w + vr + 8 * half;
            kf16[(((size_t)(b * Nn + m)) * Nn + n) * Dn + col] = (_Float16)(acc[nt][vr] + bkc);
        }
    }
}

// ---------------- scores: out/in q.k dots ----------------
__global__ void scores_kernel(const _Float16* __restrict__ kf16, const float* __restrict__ q,
                              float* __restrict__ outS, float* __restrict__ inS) {
    __shared__ float qm[DKn];
    int h = blockIdx.x, m = blockIdx.y, b = blockIdx.z;
    int n = threadIdx.x;  // 128
    if (n < DKn) qm[n] = q[(((size_t)b * Hn + h) * Nn + m) * DKn + n];
    __syncthreads();
    const h8* kp = (const h8*)(kf16 + (((size_t)(b * Nn + m)) * Nn + n) * Dn + h * DKn);
    const float* qn = q + (((size_t)b * Hn + h) * Nn + n) * DKn;
    float so = 0.f, si = 0.f;
#pragma unroll
    for (int g = 0; g < 4; ++g) {
        h8 kv = kp[g];
#pragma unroll
        for (int j = 0; j < 8; ++j) {
            float kd = (float)kv[j];
            so += qm[g * 8 + j] * kd;
            si += qn[g * 8 + j] * kd;
        }
    }
    const float scale = 0.17677669529663687f;  // 1/sqrt(32)
    size_t ro = ((size_t)b * Hn + h) * Nn;
    outS[(ro + m) * Nn + n] = so * scale;
    inS [(ro + n) * Nn + m] = si * scale;
}

// ---------------- fused dual softmax + diag fix + adj scale -> message ----------------
__global__ void softmax_msg_kernel(const float* __restrict__ outS, const float* __restrict__ inS,
                                   const float* __restrict__ adjs, const int* __restrict__ mask,
                                   float* __restrict__ message) {
    __shared__ float red[Nn];
    int i = blockIdx.x, h = blockIdx.y, b = blockIdx.z;
    int j = threadIdx.x;
    size_t ro = (((size_t)b * Hn + h) * Nn + i) * Nn;
    bool mk = mask[b * Nn + j] != 0;
    float os = mk ? outS[ro + j] : -__builtin_inff();
    float is = mk ? inS [ro + j] : -__builtin_inff();
    float mo = blockMax128(os, red, j);
    float eo = __expf(os - mo);
    float so = blockSum128(eo, red, j);
    float mi = blockMax128(is, red, j);
    float ei = __expf(is - mi);
    float si = blockSum128(ei, red, j);
    float oa = eo / so, ia = ei / si;
    float msg = (j == i) ? ia : (oa + ia);  // diag: out+in-diag(out) = in
    message[ro + j] = msg * adjs[((size_t)b * Nn + i) * Nn + j];
}

// ---------------- node_hidden = message @ v, stored (b,m,D) ----------------
__global__ void node_hidden_kernel(const float* __restrict__ message, const float* __restrict__ v,
                                   float* __restrict__ nhx) {
    __shared__ float mrow[Nn];
    int m = blockIdx.x, h = blockIdx.y, b = blockIdx.z;
    int d = threadIdx.x;  // 32
    size_t ro = (((size_t)b * Hn + h) * Nn + m) * Nn;
    for (int t = d; t < Nn; t += DKn) mrow[t] = message[ro + t];
    __syncthreads();
    float acc = 0.f;
    const float* vb = v + ((size_t)b * Hn + h) * Nn * DKn + d;
    for (int n = 0; n < Nn; ++n) acc += mrow[n] * vb[(size_t)n * DKn];
    nhx[((size_t)(b * Nn + m)) * Dn + h * DKn + d] = acc;
}

// ---------------- x = nhx @ Wo + bo ----------------
__global__ void out_proj_kernel(const float* __restrict__ nhx, const float* __restrict__ W,
                                const float* __restrict__ bias, float* __restrict__ xo) {
    __shared__ float row[Dn];
    int rid = blockIdx.x, tid = threadIdx.x;
    row[tid]       = nhx[(size_t)rid * Dn + tid];
    row[tid + 256] = nhx[(size_t)rid * Dn + tid + 256];
    __syncthreads();
    for (int cc = 0; cc < 2; ++cc) {
        int c = tid + cc * 256;
        float acc = bias[c];
        for (int k = 0; k < Dn; ++k) acc += row[k] * W[(size_t)k * Dn + c];
        xo[(size_t)rid * Dn + c] = acc;
    }
}

// ---------------- big GEMM 2: e = (message .* k) @ We + be ----------------
__global__ __launch_bounds__(256) void e_gemm_kernel(const _Float16* __restrict__ kf16,
                                                     const _Float16* __restrict__ WtE,
                                                     const float* __restrict__ be,
                                                     const float* __restrict__ message,
                                                     float* __restrict__ eOut) {
    __shared__ __align__(16) _Float16 sA[2][Nn * 64];
    __shared__ __align__(16) _Float16 sB[2][Nn * 64];
    int nb = blockIdx.x, m = blockIdx.y, b = blockIdx.z;
    int tid = threadIdx.x;
    int w = tid >> 5, lane = tid & 31, r = lane & 15, half = lane >> 4;

    v8f acc[8] = {};
    const _Float16* Abase = kf16 + ((size_t)(b * Nn + m)) * Nn * Dn;
    const _Float16* Bbase = WtE + (size_t)nb * 128 * Dn;

    h8 aReg[4], bReg[4];
    float mReg[4];

    // prologue: chunk 0
#pragma unroll
    for (int i = 0; i < 4; ++i) {
        int idx = tid + i * 256, n = idx >> 3, k8 = (idx & 7) << 3;
        aReg[i] = *(const h8*)(Abase + (size_t)n * Dn + k8);
        int h = k8 >> 5;
        mReg[i] = message[(((size_t)b * Hn + h) * Nn + m) * Nn + n];
        bReg[i] = *(const h8*)(Bbase + (size_t)n * Dn + k8);
    }
#pragma unroll
    for (int i = 0; i < 4; ++i) {
        int idx = tid + i * 256, n = idx >> 3, k8 = (idx & 7) << 3;
        *(h8*)&sA[0][n * 64 + k8] = aReg[i] * (_Float16)mReg[i];
        *(h8*)&sB[0][n * 64 + k8] = bReg[i];
    }
    __syncthreads();

    int buf = 0;
    for (int c = 0; c < 8; ++c) {
        int kn = (c + 1) * 64;
        if (c < 7) {
#pragma unroll
            for (int i = 0; i < 4; ++i) {
                int idx = tid + i * 256, n = idx >> 3, k8 = (idx & 7) << 3;
                aReg[i] = *(const h8*)(Abase + (size_t)n * Dn + kn + k8);
                int h = (kn + k8) >> 5;
                mReg[i] = message[(((size_t)b * Hn + h) * Nn + m) * Nn + n];
                bReg[i] = *(const h8*)(Bbase + (size_t)n * Dn + kn + k8);
            }
        }
#pragma unroll
        for (int kk = 0; kk < 64; kk += 32) {
            v16h af = load_afrag(&sA[buf][(16 * w + r) * 64 + kk + half * 8]);
#pragma unroll
            for (int g = 0; g < 2; ++g) {
                v16h bf[4];
#pragma unroll
                for (int t = 0; t < 4; ++t)
                    bf[t] = load_bfrag(&sB[buf][(16 * (g * 4 + t) + r) * 64 + kk + half * 8]);
#pragma unroll
                for (int t = 0; t < 4; ++t)
                    acc[g * 4 + t] = __builtin_amdgcn_wmma_f32_16x16x32_f16(
                        false, af, false, bf[t], (short)0, acc[g * 4 + t], false, false);
            }
        }
        if (c < 7) {
#pragma unroll
            for (int i = 0; i < 4; ++i) {
                int idx = tid + i * 256, n = idx >> 3, k8 = (idx & 7) << 3;
                *(h8*)&sA[buf ^ 1][n * 64 + k8] = aReg[i] * (_Float16)mReg[i];
                *(h8*)&sB[buf ^ 1][n * 64 + k8] = bReg[i];
            }
        }
        __syncthreads();
        buf ^= 1;
    }
    float* ebase = eOut + ((size_t)(b * Nn + m)) * Nn * Dn;
#pragma unroll
    for (int nt = 0; nt < 8; ++nt) {
        int col = nb * 128 + nt * 16 + r;
        float bec = be[col];
#pragma unroll
        for (int vr = 0; vr < 8; ++vr) {
            int n = 16 * w + vr + 8 * half;
            ebase[(size_t)n * Dn + col] = acc[nt][vr] + bec;
        }
    }
}

// ---------------- launch ----------------
extern "C" void kernel_launch(void* const* d_in, const int* in_sizes, int n_in,
                              void* d_out, int out_size, void* d_ws, size_t ws_size,
                              hipStream_t stream) {
    const float* query_node = (const float*)d_in[0];
    const float* value_node = (const float*)d_in[1];
    const float* key_edge   = (const float*)d_in[2];
    const float* adj_matrix = (const float*)d_in[3];
    const int*   mask       = (const int*)d_in[4];
    const float* Wq = (const float*)d_in[5];
    const float* bq = (const float*)d_in[6];
    const float* Wk = (const float*)d_in[7];
    const float* bk = (const float*)d_in[8];
    const float* Wv = (const float*)d_in[9];
    const float* bv = (const float*)d_in[10];
    const float* Wo = (const float*)d_in[11];
    const float* bo = (const float*)d_in[12];
    const float* We = (const float*)d_in[13];
    const float* be = (const float*)d_in[14];
    const float* lam = (const float*)d_in[15];

    char* ws = (char*)d_ws;
    _Float16* WtK   = (_Float16*)(ws + 0);
    _Float16* WtE   = (_Float16*)(ws + 524288);
    float* qbuf     = (float*)(ws + 1048576);
    float* vbuf     = (float*)(ws + 3145728);
    float* adjs     = (float*)(ws + 5242880);
    float* nhx      = (float*)(ws + 5767168);
    float* outS     = (float*)(ws + 7864320);
    float* inS      = (float*)(ws + 16252928);
    float* message  = (float*)(ws + 24641536);
    _Float16* kf16  = (_Float16*)(ws + 33030144);  // needs total ws >= 167,247,872 B

    float* xo = (float*)d_out;                         // (B,N,D)
    float* eo = (float*)d_out + (size_t)Bn * Nn * Dn;  // (B,N,N,D)

    wconv_kernel<<<dim3(32, 32), dim3(16, 16), 0, stream>>>(Wk, WtK);
    wconv_kernel<<<dim3(32, 32), dim3(16, 16), 0, stream>>>(We, WtE);
    qv_proj_kernel<<<Bn * Nn, 256, 0, stream>>>(query_node, Wq, bq, qbuf);
    qv_proj_kernel<<<Bn * Nn, 256, 0, stream>>>(value_node, Wv, bv, vbuf);
    adj_softmax_kernel<<<dim3(Nn, Bn), Nn, 0, stream>>>(adj_matrix, mask, lam, adjs);

    k_gemm_kernel<<<dim3(4, Nn, Bn), 256, 0, stream>>>(key_edge, WtK, bk, kf16);
    scores_kernel<<<dim3(Hn, Nn, Bn), Nn, 0, stream>>>(kf16, qbuf, outS, inS);
    softmax_msg_kernel<<<dim3(Nn, Hn, Bn), Nn, 0, stream>>>(outS, inS, adjs, mask, message);
    node_hidden_kernel<<<dim3(Nn, Hn, Bn), DKn, 0, stream>>>(message, vbuf, nhx);
    out_proj_kernel<<<Bn * Nn, 256, 0, stream>>>(nhx, Wo, bo, xo);
    e_gemm_kernel<<<dim3(4, Nn, Bn), 256, 0, stream>>>(kf16, WtE, be, message, eo);
}
